// LuongAttention_56710748176621
// MI455X (gfx1250) — compile-verified, matrix operand order checked by reading, not run on hardware
//
#include <hip/hip_runtime.h>

// ---------------- types ----------------
typedef __bf16 bf16_t;
typedef __attribute__((ext_vector_type(16))) __bf16 v16bf;
typedef __attribute__((ext_vector_type(8)))  __bf16 v8bf;
typedef __attribute__((ext_vector_type(4)))  __bf16 v4bf;
typedef __attribute__((ext_vector_type(8)))  float  v8f;
typedef __attribute__((ext_vector_type(4)))  float  v4f;

// ---------------- problem constants ----------------
static constexpr int kB   = 16;
static constexpr int kTQ  = 2048;
static constexpr int kTV  = 2048;
static constexpr int kDEC = 1024;
static constexpr int kENC = 512;

// LDS layout for fused attention kernel
static constexpr int kSLD  = kTV + 4;  // padded f32 score row (bank-conflict avoidance)
static constexpr int kVTLD = 48;       // padded halves per transposed-value row (96B, 16B aligned)

// ---------------- helpers ----------------
__device__ __forceinline__ bf16_t f2bf(float f) {
  union { float f; unsigned u; } v; v.f = f;
  unsigned r = v.u + 0x7FFFu + ((v.u >> 16) & 1u);   // round-to-nearest-even
  unsigned short h = (unsigned short)(r >> 16);
  return __builtin_bit_cast(bf16_t, h);
}

__device__ __forceinline__ v8f wmma_bf16(v16bf a, v16bf b, v8f c) {
  // v_wmma_f32_16x16x32_bf16: D = A(16x32) * B(32x16) + C
  return __builtin_amdgcn_wmma_f32_16x16x32_bf16(false, a, false, b, (short)0, c, false, false);
}

// Load a 16x32 bf16 fragment per ISA 7.12.2:
//   lane<16 : row = lane,    halves [kc..kc+7] -> elems 0..7, [kc+16..kc+23] -> elems 8..15
//   lane>=16: row = lane-16, halves [kc+8..kc+15] / [kc+24..kc+31]
__device__ __forceinline__ v16bf load_frag(const bf16_t* __restrict__ base, int ld, int kc) {
  int lane = threadIdx.x & 31;
  int row  = lane & 15;
  int koff = (lane >> 4) << 3;
  const bf16_t* p = base + (size_t)row * ld + kc + koff;
  v8bf lo = *(const v8bf*)p;
  v8bf hi = *(const v8bf*)(p + 16);
  return __builtin_shufflevector(lo, hi, 0,1,2,3,4,5,6,7,8,9,10,11,12,13,14,15);
}

// ---------------- kernel 0: f32 -> bf16 conversions ----------------
__global__ void cvt_to_bf16(const float* __restrict__ in, bf16_t* __restrict__ out, int n4) {
  int i = blockIdx.x * blockDim.x + threadIdx.x;
  int stride = gridDim.x * blockDim.x;
  for (; i < n4; i += stride) {
    v4f x = ((const v4f*)in)[i];
    v4bf y;
    y[0] = f2bf(x[0]); y[1] = f2bf(x[1]); y[2] = f2bf(x[2]); y[3] = f2bf(x[3]);
    ((v4bf*)out)[i] = y;
  }
}

// W [ENC, DEC] -> wT bf16 [DEC, ENC]
__global__ void cvt_w_transpose(const float* __restrict__ W, bf16_t* __restrict__ wT) {
  int i = blockIdx.x * blockDim.x + threadIdx.x;
  if (i < kENC * kDEC) {
    int e = i / kDEC;
    int d = i % kDEC;
    wT[(size_t)d * kENC + e] = f2bf(W[i]);
  }
}

// ---------------- kernel 1: keys = value @ W + b  (bf16 out) ----------------
// One 16x64 strip per wave (A-frag reused across 4 B tiles).
// rows = B*TV = 32768 (2048 tiles), cols = DEC/64 = 16 col-groups.
__global__ __launch_bounds__(256)
void keys_gemm(const bf16_t* __restrict__ vbf, const bf16_t* __restrict__ wT,
               const float* __restrict__ bias, bf16_t* __restrict__ keys) {
  int wave = (blockIdx.x * blockDim.x + threadIdx.x) >> 5;
  int lane = threadIdx.x & 31;
  int r0 = (wave >> 4) << 4;   // row-tile * 16
  int d0 = (wave & 15) << 6;   // col-group * 64

  const bf16_t* arow = vbf + (size_t)r0 * kENC;
  v8f acc[4] = {};
  for (int kc = 0; kc < kENC; kc += 32) {
    v16bf a = load_frag(arow, kENC, kc);
#pragma unroll
    for (int t = 0; t < 4; ++t) {
      v16bf b = load_frag(wT + (size_t)(d0 + t * 16) * kENC, kENC, kc);
      acc[t] = wmma_bf16(a, b, acc[t]);
    }
  }
  int n = lane & 15;
  int mbase = (lane >> 4) << 3;
#pragma unroll
  for (int t = 0; t < 4; ++t) {
    float bb = bias[d0 + t * 16 + n];
#pragma unroll
    for (int r = 0; r < 8; ++r) {
      int m = mbase + r;
      keys[(size_t)(r0 + m) * kDEC + d0 + t * 16 + n] = f2bf(acc[t][r] + bb);
    }
  }
}

// ---------------- kernel 2: fused scores -> softmax -> alignment_t + context ----------------
// One workgroup = one batch, 16 query rows.  8 waves.
__global__ __launch_bounds__(256, 1)
void attn_fused(const bf16_t* __restrict__ qbf, const bf16_t* __restrict__ keys,
                const bf16_t* __restrict__ vbf,
                float* __restrict__ out_ctx, float* __restrict__ out_align) {
  __shared__ float  s_scores[16 * kSLD];      // 16 x 2052 f32 = 131,328 B
  __shared__ bf16_t s_vT[kENC * kVTLD];       // 512 x 48 bf16 =  49,152 B

  int b    = blockIdx.x >> 7;                 // 128 q-tiles per batch
  int q0   = (blockIdx.x & 127) << 4;
  int wid  = threadIdx.x >> 5;
  int lane = threadIdx.x & 31;

  const bf16_t* qbase = qbf  + ((size_t)b * kTQ + q0) * kDEC;
  const bf16_t* kbase = keys + (size_t)b * kTV * kDEC;

  // ---- Phase A: scores[16][2048] = q_tile @ keys^T ----
  // Each wave owns a 16x128 strip (8 accumulator tiles); 2 strips per wave.
#pragma unroll 1
  for (int g = 0; g < 2; ++g) {
    int vc0 = wid * 256 + g * 128;
    const bf16_t* krow0 = kbase + (size_t)vc0 * kDEC;
    __builtin_prefetch(krow0, 0, 1);           // global_prefetch
    v8f acc[8] = {};
    for (int kc = 0; kc < kDEC; kc += 32) {
      v16bf a = load_frag(qbase, kDEC, kc);    // A-frag reused across 8 WMMAs
#pragma unroll
      for (int t = 0; t < 8; ++t) {
        v16bf bm = load_frag(krow0 + (size_t)(t * 16) * kDEC, kDEC, kc);
        acc[t] = wmma_bf16(a, bm, acc[t]);
      }
    }
    int n = lane & 15;
    int mbase = (lane >> 4) << 3;
#pragma unroll
    for (int t = 0; t < 8; ++t)
#pragma unroll
      for (int r = 0; r < 8; ++r)
        s_scores[(mbase + r) * kSLD + vc0 + t * 16 + n] = acc[t][r];
  }
  __syncthreads();

  // ---- softmax: each wave owns 2 rows ----
  for (int m = wid * 2; m < wid * 2 + 2; ++m) {
    float* row = s_scores + m * kSLD;
    float mx = -3.0e38f;
    for (int v = lane; v < kTV; v += 32) mx = fmaxf(mx, row[v]);
#pragma unroll
    for (int o = 16; o > 0; o >>= 1) mx = fmaxf(mx, __shfl_xor(mx, o, 32));
    float sum = 0.f;
    for (int v = lane; v < kTV; v += 32) {
      float e = __expf(row[v] - mx);
      row[v] = e;
      sum += e;
    }
#pragma unroll
    for (int o = 16; o > 0; o >>= 1) sum += __shfl_xor(sum, o, 32);
    float inv = 1.0f / sum;
    for (int v = lane; v < kTV; v += 32) row[v] *= inv;
  }
  __syncthreads();

  // ---- write alignment_t: out_align[b][v][q0..q0+15] ----
  {
    float* oa = out_align + (size_t)b * kTV * kTQ + q0;
    for (int v = threadIdx.x; v < kTV; v += 256) {
      float* dst = oa + (size_t)v * kTQ;
#pragma unroll
      for (int m = 0; m < 16; m += 4) {
        v4f t = { s_scores[(m + 0) * kSLD + v], s_scores[(m + 1) * kSLD + v],
                  s_scores[(m + 2) * kSLD + v], s_scores[(m + 3) * kSLD + v] };
        *(v4f*)(dst + m) = t;
      }
    }
  }

  // ---- Phase C: context[16][512] = P @ value.  wave wid -> cols [wid*64, wid*64+64) ----
  v8f acc[4] = {};
  int e0 = wid * 64;
  const bf16_t* vb = vbf + (size_t)b * kTV * kENC;
  for (int kc = 0; kc < kTV; kc += 32) {
    __syncthreads();
    // stage value[kc..kc+31][0..511] transposed into s_vT[e][k]
    const bf16_t* vsrc = vb + (size_t)kc * kENC;
    for (int idx = threadIdx.x * 8; idx < 32 * kENC; idx += 256 * 8) {
      int k = idx / kENC;
      int e = idx % kENC;
      v8bf t = *(const v8bf*)(vsrc + (size_t)k * kENC + e);
#pragma unroll
      for (int j = 0; j < 8; ++j) s_vT[(e + j) * kVTLD + k] = t[j];
    }
    __syncthreads();

    // A-frag from normalized probabilities (f32 LDS -> bf16)
    int row  = lane & 15;
    int koff = (lane >> 4) << 3;
    const float* sr = s_scores + row * kSLD + kc + koff;
    v16bf a;
#pragma unroll
    for (int j = 0; j < 8; ++j) a[j]     = f2bf(sr[j]);
#pragma unroll
    for (int j = 0; j < 8; ++j) a[8 + j] = f2bf(sr[16 + j]);

#pragma unroll
    for (int t = 0; t < 4; ++t) {
      int ecol = e0 + t * 16 + (lane & 15);
      const bf16_t* bp = s_vT + (size_t)ecol * kVTLD + koff;
      v8bf lo = *(const v8bf*)bp;
      v8bf hi = *(const v8bf*)(bp + 16);
      v16bf bm = __builtin_shufflevector(lo, hi, 0,1,2,3,4,5,6,7,8,9,10,11,12,13,14,15);
      acc[t] = wmma_bf16(a, bm, acc[t]);
    }
  }

  // store context part (first 512 of 1536 cols)
  float* oc = out_ctx + ((size_t)b * kTQ + q0) * (kENC + kDEC);
  int n = lane & 15;
  int mb = (lane >> 4) << 3;
#pragma unroll
  for (int t = 0; t < 4; ++t)
#pragma unroll
    for (int r = 0; r < 8; ++r)
      oc[(size_t)(mb + r) * (kENC + kDEC) + e0 + t * 16 + n] = acc[t][r];
}

// ---------------- kernel 3: concat query into context[:, 512:1536] ----------------
__global__ void concat_query(const float* __restrict__ q, float* __restrict__ out_ctx) {
  size_t total4 = (size_t)kB * kTQ * kDEC / 4;
  size_t i = (size_t)blockIdx.x * blockDim.x + threadIdx.x;
  size_t stride = (size_t)gridDim.x * blockDim.x;
  for (; i < total4; i += stride) {
    v4f x = ((const v4f*)q)[i];
    size_t el  = i * 4;
    size_t row = el / kDEC;
    size_t col = el % kDEC;
    *(v4f*)(out_ctx + row * (size_t)(kENC + kDEC) + kENC + col) = x;
  }
}

// ---------------- host launcher ----------------
extern "C" void kernel_launch(void* const* d_in, const int* in_sizes, int n_in,
                              void* d_out, int out_size, void* d_ws, size_t ws_size,
                              hipStream_t stream) {
  const float* query = (const float*)d_in[0];   // [16,2048,1024]
  const float* value = (const float*)d_in[1];   // [16,2048,512]
  const float* W     = (const float*)d_in[2];   // [512,1024]
  const float* bias  = (const float*)d_in[3];   // [1024]

  // workspace layout (bytes)
  char* ws = (char*)d_ws;
  size_t off = 0;
  bf16_t* qbf  = (bf16_t*)(ws + off); off += (size_t)kB * kTQ * kDEC * 2;   // 64 MB
  bf16_t* vbf  = (bf16_t*)(ws + off); off += (size_t)kB * kTV * kENC * 2;   // 32 MB
  bf16_t* wT   = (bf16_t*)(ws + off); off += (size_t)kENC * kDEC * 2;       //  1 MB
  bf16_t* keys = (bf16_t*)(ws + off); off += (size_t)kB * kTV * kDEC * 2;   // 64 MB
  (void)ws_size; (void)in_sizes; (void)n_in; (void)out_size;

  float* out_ctx   = (float*)d_out;
  float* out_align = out_ctx + (size_t)kB * kTQ * (kENC + kDEC);

  // 0) conversions
  cvt_to_bf16<<<4096, 256, 0, stream>>>(query, qbf, kB * kTQ * kDEC / 4);
  cvt_to_bf16<<<4096, 256, 0, stream>>>(value, vbf, kB * kTV * kENC / 4);
  cvt_w_transpose<<<(kENC * kDEC) / 256, 256, 0, stream>>>(W, wT);

  // 1) keys GEMM: 2048 row-tiles * 16 col-groups = 32768 waves, 8 waves/block
  keys_gemm<<<(2048 * 16) / 8, 256, 0, stream>>>(vbf, wT, bias, keys);

  // 2) fused attention: one block per (batch, 16-query tile)
  attn_fused<<<kB * (kTQ / 16), 256, 0, stream>>>(qbf, keys, vbf, out_ctx, out_align);

  // 3) concat query into context
  concat_query<<<4096, 256, 0, stream>>>(query, out_ctx);
}
